// RNNLanguageModel_76184129897056
// MI455X (gfx1250) — matrix operand order repaired and checked by reading, not compiled, not run on hardware
//
#include <hip/hip_runtime.h>
#include <hip/hip_bf16.h>

// ---------------------------------------------------------------------------
// GRU language model, persistent WMMA kernel for gfx1250 (MI455X).
//   B=64, L=512 (511 steps), EMB=64, H=1024, GATES=3072, VOCAB=128
// Strategy: sequential chain -> one persistent kernel, 2 grid barriers/step;
// bf16 WMMA everywhere; weights stream from L2 (20MB bf16 << 192MB L2);
// shared A-tiles staged into LDS via async global->LDS copies.
// ---------------------------------------------------------------------------

#define NB 64      // batch
#define NL 512     // sequence length
#define NT 511     // time steps (L-1)
#define NE 64      // embedding dim
#define NH 1024    // hidden dim
#define NV 128     // vocab
#define NWG 32     // persistent workgroups
#define NTHR 256   // threads per workgroup (8 waves of 32)

typedef __attribute__((ext_vector_type(16))) __bf16 v16bf;
typedef __attribute__((ext_vector_type(8)))  float  v8f;
typedef __attribute__((ext_vector_type(4)))  int    v4i_t;

// ---------------------------------------------------------------------------
// async global -> LDS copy (CDNA5), with synchronous fallback
// builtin signature (from clang diagnostic): (global int4*, lds int4*, Ii, Ii)
// ---------------------------------------------------------------------------
#if __has_builtin(__builtin_amdgcn_global_load_async_to_lds_b128)
#define ASYNC_LDS 1
typedef __attribute__((address_space(1))) v4i_t g_v4i;
typedef __attribute__((address_space(3))) v4i_t l_v4i;
#else
#define ASYNC_LDS 0
#endif

__device__ __forceinline__ void async_cp16(const char* g, char* l) {
#if ASYNC_LDS
  __builtin_amdgcn_global_load_async_to_lds_b128((g_v4i*)g, (l_v4i*)l, 0, 0);
#else
  *(uint4*)l = *(const uint4*)g;
#endif
}

__device__ __forceinline__ void async_wait() {
#if ASYNC_LDS
#if __has_builtin(__builtin_amdgcn_s_wait_asynccnt)
  __builtin_amdgcn_s_wait_asynccnt(0);
#else
  asm volatile("s_wait_asynccnt 0x0" ::: "memory");
#endif
#endif
}

// stage a contiguous block of bytes (multiple of 16) into LDS, whole block
__device__ __forceinline__ void stage_bytes(const char* g, char* l, int nbytes) {
  for (int off = threadIdx.x * 16; off < nbytes; off += NTHR * 16)
    async_cp16(g + off, l + off);
}

// ---------------------------------------------------------------------------
// WMMA helpers
// ---------------------------------------------------------------------------
__device__ __forceinline__ v16bf load_tile(const __bf16* p0, const __bf16* p1) {
  union { uint4 q[2]; v16bf v; } u;
  u.q[0] = *(const uint4*)p0;   // global_load_b128 / ds_load_b128
  u.q[1] = *(const uint4*)p1;
  return u.v;
}

__device__ __forceinline__ v8f wmma_bf16(v16bf a, v16bf b, v8f c) {
  return __builtin_amdgcn_wmma_f32_16x16x32_bf16(false, a, false, b,
                                                 (short)0, c, false, false);
}

__device__ __forceinline__ float sigmoidf_(float x) {
  x = fminf(fmaxf(x, -30.f), 30.f);
  return 1.f / (1.f + __expf(-x));
}

__device__ __forceinline__ float tanhf_(float x) {
  x = fminf(fmaxf(x, -15.f), 15.f);
  float e = __expf(-2.f * x);
  return (1.f - e) / (1.f + e);
}

// sense-free generation barrier across the persistent grid
__device__ __forceinline__ void grid_barrier(unsigned* cnt, unsigned* gen) {
  __syncthreads();
  if (threadIdx.x == 0) {
    __threadfence();
    unsigned g = __hip_atomic_load(gen, __ATOMIC_RELAXED, __HIP_MEMORY_SCOPE_AGENT);
    if (__hip_atomic_fetch_add(cnt, 1u, __ATOMIC_ACQ_REL, __HIP_MEMORY_SCOPE_AGENT)
        == (unsigned)(NWG - 1)) {
      __hip_atomic_store(cnt, 0u, __ATOMIC_RELAXED, __HIP_MEMORY_SCOPE_AGENT);
      __hip_atomic_fetch_add(gen, 1u, __ATOMIC_ACQ_REL, __HIP_MEMORY_SCOPE_AGENT);
    } else {
      while (__hip_atomic_load(gen, __ATOMIC_ACQUIRE, __HIP_MEMORY_SCOPE_AGENT) == g)
        __builtin_amdgcn_s_sleep(2);
    }
  }
  __syncthreads();
}

// ---------------------------------------------------------------------------
// one GRU output tile (rows 16*mtile.., cols 16*jtile..) with A staged in LDS.
// sAx: 16 x ldx input block; sAh: 16 x NH hidden block (== h_old rows).
// ---------------------------------------------------------------------------
__device__ __forceinline__ void gru_layer_tile_lds(
    const __bf16* sAx, int ldx, int kx_tiles,
    const __bf16* sAh,
    const __bf16* __restrict__ Wih, int ldi,
    const __bf16* __restrict__ Whh,
    float br_b, float bz_b, float bin_b, float bhn_b,
    __bf16* __restrict__ hnew,
    int mtile, int jtile, int lane)
{
  const int l16 = lane & 15, lh = lane >> 4;
  const int jr = jtile * 16 + l16;
  v8f accr = {}, accz = {}, accin = {}, acchn = {};

  const __bf16* axr = sAx + l16 * ldx;      // LDS
  const __bf16* ahr = sAh + l16 * NH;       // LDS
  const __bf16* wir = Wih + (size_t)jr * ldi;
  const __bf16* wiz = Wih + (size_t)(NH + jr) * ldi;
  const __bf16* win = Wih + (size_t)(2 * NH + jr) * ldi;
  const __bf16* whr = Whh + (size_t)jr * NH;
  const __bf16* whz = Whh + (size_t)(NH + jr) * NH;
  const __bf16* whn = Whh + (size_t)(2 * NH + jr) * NH;

  for (int kb = 0; kb < kx_tiles; ++kb) {
    const int ka = kb * 32 + 8 * lh;
    const int kw = kb * 32 + 8 * lh;
    v16bf a  = load_tile(axr + ka, axr + ka + 16);
    v16bf br = load_tile(wir + kw, wir + kw + 16);
    v16bf bz = load_tile(wiz + kw, wiz + kw + 16);
    v16bf bn = load_tile(win + kw, win + kw + 16);
    accr  = wmma_bf16(a, br, accr);
    accz  = wmma_bf16(a, bz, accz);
    accin = wmma_bf16(a, bn, accin);
  }
  for (int kb = 0; kb < NH / 32; ++kb) {
    const int ka = kb * 32 + 8 * lh;
    const int kw = kb * 32 + 8 * lh;
    v16bf a  = load_tile(ahr + ka, ahr + ka + 16);
    v16bf br = load_tile(whr + kw, whr + kw + 16);
    v16bf bz = load_tile(whz + kw, whz + kw + 16);
    v16bf bn = load_tile(whn + kw, whn + kw + 16);
    accr  = wmma_bf16(a, br, accr);
    accz  = wmma_bf16(a, bz, accz);
    acchn = wmma_bf16(a, bn, acchn);
  }

#pragma unroll
  for (int i = 0; i < 8; ++i) {
    const int row = lh * 8 + i;                  // C/D layout: M = i + 8*(l/16)
    float r = sigmoidf_(accr[i] + br_b);
    float z = sigmoidf_(accz[i] + bz_b);
    float n = tanhf_(accin[i] + bin_b + r * (acchn[i] + bhn_b));
    float ho = (float)sAh[row * NH + jr];        // staged h_old (LDS)
    const int b = mtile * 16 + row;
    hnew[(size_t)b * NH + jr] = (__bf16)((1.f - z) * n + z * ho);
  }
}

// ---------------------------------------------------------------------------
// persistent kernel
// ---------------------------------------------------------------------------
__launch_bounds__(NTHR, 1)
__global__ void gru_lm_kernel(
    const __bf16* __restrict__ X,          // [NT][NB][NE]
    const __bf16* __restrict__ Wih0b, const __bf16* __restrict__ Whh0b,
    const __bf16* __restrict__ Wih1b, const __bf16* __restrict__ Whh1b,
    const __bf16* __restrict__ Woutb,
    const float* __restrict__ bih0, const float* __restrict__ bhh0,
    const float* __restrict__ bih1, const float* __restrict__ bhh1,
    const float* __restrict__ bout,
    const int*  __restrict__ batch,
    __bf16* __restrict__ h0,               // [2][NB][NH] ping-pong
    __bf16* __restrict__ h1,               // [2][NB][NH] ping-pong
    float* __restrict__ out,               // [NB][NT]
    unsigned* __restrict__ bar)
{
  // 64KB LDS arena, time-multiplexed:
  //  layer0: x(2KB)@0,  h0(32KB)@2048
  //  layer1: h0new(32KB)@0, h1(32KB)@32768
  //  logits: h1new(32KB)@0,  slog(8KB)@32768
  __shared__ __align__(16) char smem[65536];
  __bf16* sm_bf = (__bf16*)smem;

  const int lane = threadIdx.x & 31;
  const int wave = threadIdx.x >> 5;
  const int bid = blockIdx.x;
  const int mtile = bid >> 3;                        // 0..3 (block-uniform)
  const int jtile = (bid & 7) * 8 + wave;            // 0..63
  unsigned* cnt = bar;
  unsigned* gen = bar + 16;

  // hoist time-invariant biases
  const int l16 = lane & 15, lh = lane >> 4;
  const int jr = jtile * 16 + l16;
  const float b0r = bih0[jr] + bhh0[jr];
  const float b0z = bih0[NH + jr] + bhh0[NH + jr];
  const float b0i = bih0[2 * NH + jr];
  const float b0h = bhh0[2 * NH + jr];
  const float b1r = bih1[jr] + bhh1[jr];
  const float b1z = bih1[NH + jr] + bhh1[NH + jr];
  const float b1i = bih1[2 * NH + jr];
  const float b1h = bhh1[2 * NH + jr];
  const float bo  = (bid < 4) ? bout[wave * 16 + l16] : 0.f;

  const size_t HS = (size_t)NB * NH;
  const size_t rowoff = (size_t)mtile * 16 * NH;     // block's A-row base

  for (int t = 0; t < NT; ++t) {
    const int p = t & 1;
    const __bf16* h0r = h0 + (size_t)p * HS;
    __bf16*       h0w = h0 + (size_t)(1 - p) * HS;
    const __bf16* h1r = h1 + (size_t)p * HS;
    __bf16*       h1w = h1 + (size_t)(1 - p) * HS;

    // ---- stage layer0 A blocks: x_t rows + h0 rows (contiguous) ----
    __syncthreads();                                  // smem reuse guard
    stage_bytes((const char*)(X + ((size_t)t * NB + mtile * 16) * NE),
                smem, 16 * NE * 2);
    stage_bytes((const char*)(h0r + rowoff), smem + 2048, 16 * NH * 2);
    async_wait();
    __syncthreads();

    // ---- layer 0: x_t (K=64) + h0 (K=1024) ----
    gru_layer_tile_lds(sm_bf, NE, NE / 32, (__bf16*)(smem + 2048),
                       Wih0b, NE, Whh0b, b0r, b0z, b0i, b0h,
                       h0w, mtile, jtile, lane);
    grid_barrier(cnt, gen);

    // ---- stage layer1 A blocks: h0_new rows + h1 rows ----
    stage_bytes((const char*)(h0w + rowoff), smem, 16 * NH * 2);
    stage_bytes((const char*)(h1r + rowoff), smem + 32768, 16 * NH * 2);
    async_wait();
    __syncthreads();

    // ---- layer 1: h0_new (K=1024) + h1 (K=1024) ----
    gru_layer_tile_lds(sm_bf, NH, NH / 32, (__bf16*)(smem + 32768),
                       Wih1b, NH, Whh1b, b1r, b1z, b1i, b1h,
                       h1w, mtile, jtile, lane);
    grid_barrier(cnt, gen);

    // ---- logits + log-softmax (blocks 0..3 own one 16-row M-tile) ----
    if (bid < 4) {
      stage_bytes((const char*)(h1w + (size_t)bid * 16 * NH), smem, 16 * NH * 2);
      async_wait();
      __syncthreads();

      float* slog = (float*)(smem + 32768);
      v8f acc = {};
      const __bf16* ahr  = sm_bf + l16 * NH;                    // LDS
      const __bf16* wrow = Woutb + (size_t)(wave * 16 + l16) * NH;
      for (int kb = 0; kb < NH / 32; ++kb) {
        const int ka = kb * 32 + 8 * lh;
        const int kw = kb * 32 + 8 * lh;
        v16bf a  = load_tile(ahr + ka, ahr + ka + 16);
        v16bf bt = load_tile(wrow + kw, wrow + kw + 16);
        acc = wmma_bf16(a, bt, acc);
      }
      const int col = wave * 16 + l16;
#pragma unroll
      for (int i = 0; i < 8; ++i)
        slog[(lh * 8 + i) * NV + col] = acc[i] + bo;
      __syncthreads();

      // row-wise log-softmax: 16 threads/row, 8 cols each, shfl reduce
      const int row = threadIdx.x >> 4;
      const int sub = threadIdx.x & 15;
      float mx = -1e30f;
#pragma unroll
      for (int c = 0; c < 8; ++c)
        mx = fmaxf(mx, slog[row * NV + sub + c * 16]);
      for (int m = 8; m >= 1; m >>= 1) mx = fmaxf(mx, __shfl_xor(mx, m, 16));
      float se = 0.f;
#pragma unroll
      for (int c = 0; c < 8; ++c)
        se += __expf(slog[row * NV + sub + c * 16] - mx);
      for (int m = 8; m >= 1; m >>= 1) se += __shfl_xor(se, m, 16);
      if (sub == 0) {
        const int b = bid * 16 + row;
        const int tok = batch[b * NL + (t + 1)];
        out[(size_t)b * NT + t] = slog[row * NV + tok] - mx - __logf(se);
      }
    }
    // next grid barrier (layer0 of t+1) covers logits/stage hazards
  }
}

// ---------------------------------------------------------------------------
// prep kernels
// ---------------------------------------------------------------------------
__global__ void f32_to_bf16_kernel(const float* __restrict__ s,
                                   __bf16* __restrict__ d, int n) {
  int i = blockIdx.x * blockDim.x + threadIdx.x;
  if (i < n) d[i] = (__bf16)s[i];
}

__global__ void embed_kernel(const int* __restrict__ batch,
                             const float* __restrict__ E,
                             __bf16* __restrict__ X) {
  int i = blockIdx.x * blockDim.x + threadIdx.x;
  if (i >= NT * NB * NE) return;
  const int k = i & (NE - 1);
  const int b = (i / NE) & (NB - 1);
  const int t = i / (NE * NB);
  const int tok = batch[b * NL + t];
  X[i] = (__bf16)E[tok * NE + k];
}

// ---------------------------------------------------------------------------
// launcher
// ---------------------------------------------------------------------------
extern "C" void kernel_launch(void* const* d_in, const int* in_sizes, int n_in,
                              void* d_out, int out_size, void* d_ws, size_t ws_size,
                              hipStream_t stream) {
  (void)in_sizes; (void)n_in; (void)out_size; (void)ws_size;
  const int*   batch = (const int*)  d_in[0];
  const float* E     = (const float*)d_in[2];
  const float* Wih0  = (const float*)d_in[3];
  const float* Whh0  = (const float*)d_in[4];
  const float* bih0  = (const float*)d_in[5];
  const float* bhh0  = (const float*)d_in[6];
  const float* Wih1  = (const float*)d_in[7];
  const float* Whh1  = (const float*)d_in[8];
  const float* bih1  = (const float*)d_in[9];
  const float* bhh1  = (const float*)d_in[10];
  const float* Wout  = (const float*)d_in[11];
  const float* bout  = (const float*)d_in[12];
  float* out = (float*)d_out;

  char* ws = (char*)d_ws;
  size_t off = 0;
  auto take = [&](size_t bytes) -> void* {
    void* p = ws + off;
    off += (bytes + 255) & ~(size_t)255;
    return p;
  };

  unsigned* bar = (unsigned*)take(256);
  __bf16* X     = (__bf16*)take((size_t)NT * NB * NE * 2);
  __bf16* Wih0b = (__bf16*)take((size_t)3 * NH * NE * 2);
  __bf16* Whh0b = (__bf16*)take((size_t)3 * NH * NH * 2);
  __bf16* Wih1b = (__bf16*)take((size_t)3 * NH * NH * 2);
  __bf16* Whh1b = (__bf16*)take((size_t)3 * NH * NH * 2);
  __bf16* Woutb = (__bf16*)take((size_t)NV * NH * 2);
  __bf16* h0    = (__bf16*)take((size_t)2 * NB * NH * 2);
  __bf16* h1    = (__bf16*)take((size_t)2 * NB * NH * 2);

  (void)hipMemsetAsync(bar, 0, 256, stream);
  (void)hipMemsetAsync(h0, 0, (size_t)2 * NB * NH * 2, stream);
  (void)hipMemsetAsync(h1, 0, (size_t)2 * NB * NH * 2, stream);

  {
    int n = NT * NB * NE;
    embed_kernel<<<(n + 255) / 256, 256, 0, stream>>>(batch, E, X);
  }
  {
    int n = 3 * NH * NE;
    f32_to_bf16_kernel<<<(n + 255) / 256, 256, 0, stream>>>(Wih0, Wih0b, n);
  }
  {
    int n = 3 * NH * NH;
    f32_to_bf16_kernel<<<(n + 255) / 256, 256, 0, stream>>>(Whh0, Whh0b, n);
    f32_to_bf16_kernel<<<(n + 255) / 256, 256, 0, stream>>>(Wih1, Wih1b, n);
    f32_to_bf16_kernel<<<(n + 255) / 256, 256, 0, stream>>>(Whh1, Whh1b, n);
  }
  {
    int n = NV * NH;
    f32_to_bf16_kernel<<<(n + 255) / 256, 256, 0, stream>>>(Wout, Woutb, n);
  }

  gru_lm_kernel<<<NWG, NTHR, 0, stream>>>(
      X, Wih0b, Whh0b, Wih1b, Whh1b, Woutb,
      bih0, bhh0, bih1, bhh1, bout,
      batch, h0, h1, out, bar);
}
